// MovieRecommendationModel_6854767805141
// MI455X (gfx1250) — matrix operand order repaired and checked by reading, not compiled, not run on hardware
//
#include <hip/hip_runtime.h>
#include <hip/hip_bf16.h>

typedef __attribute__((ext_vector_type(16))) _Float16 v16h;
typedef __attribute__((ext_vector_type(8)))  _Float16 v8h;
typedef __attribute__((ext_vector_type(8)))  float    v8f;
typedef __attribute__((address_space(3)))    _Float16 lds_f16;

#define GN  20480
#define GH  128
#define NUSERS 4096
#define NITEMS (GN - NUSERS)
#define LDS_LDA 136   // padded row stride (halves): 272B = 68 dwords -> no bank alias

// ---------------------------------------------------------------------------
// Elementwise / scatter kernels
// ---------------------------------------------------------------------------

__global__ void k_fill1(float* p, int n) {
    int i = blockIdx.x * blockDim.x + threadIdx.x;
    if (i < n) p[i] = 1.0f;   // self-loop contributes 1 to each degree
}

__global__ void k_degcount(const int* __restrict__ dst, float* __restrict__ deg, int E) {
    int e = blockIdx.x * blockDim.x + threadIdx.x;
    if (e < E) atomicAdd(&deg[dst[e]], 1.0f);
}

__global__ void k_rsqrt_inplace(float* p, int n) {
    int i = blockIdx.x * blockDim.x + threadIdx.x;
    if (i < n) {
        float d = p[i];
        p[i] = (d > 0.0f) ? rsqrtf(d) : 0.0f;
    }
}

// s[n] init = x[n]*dinv[n]  (self-loop term, final *dinv[n] applied later)
__global__ void k_s_init(const int* __restrict__ x, const float* __restrict__ dinv,
                         float* __restrict__ s, int n) {
    int i = blockIdx.x * blockDim.x + threadIdx.x;
    if (i < n) s[i] = (float)x[i] * dinv[i];
}

__global__ void k_s_scatter(const int* __restrict__ src, const int* __restrict__ dst,
                            const int* __restrict__ x, const float* __restrict__ dinv,
                            float* __restrict__ s, int E) {
    int e = blockIdx.x * blockDim.x + threadIdx.x;
    if (e < E) {
        int sn = src[e];
        atomicAdd(&s[dst[e]], (float)x[sn] * dinv[sn]);
    }
}

// h1[n][j] = relu(dinv[n]*s[n]*W1[j] + b1[j])  -> f16
__global__ void k_h1(const float* __restrict__ s, const float* __restrict__ dinv,
                     const float* __restrict__ W1, const float* __restrict__ b1,
                     _Float16* __restrict__ h1h, int n) {
    int i = blockIdx.x * blockDim.x + threadIdx.x;
    if (i < n * GH) {
        int node = i >> 7, j = i & 127;
        float v = dinv[node] * s[node] * W1[j] + b1[j];
        h1h[i] = (_Float16)(v > 0.0f ? v : 0.0f);
    }
}

// transpose + convert both weight matrices to f16 (N x K layout for WMMA B)
__global__ void k_wt(const float* __restrict__ W2, const float* __restrict__ Wl,
                     _Float16* __restrict__ W2t, _Float16* __restrict__ Wlt) {
    int i = blockIdx.x * blockDim.x + threadIdx.x;   // i = n*128 + k
    if (i < GH * GH) {
        int nn = i >> 7, k = i & 127;
        W2t[i] = (_Float16)W2[k * GH + nn];
        Wlt[i] = (_Float16)Wl[k * GH + nn];
    }
}

// acc2[n][j] = dinv[n]*z2[n][j]  (self-loop term before the final *dinv[n])
__global__ void k_acc2_init(const float* __restrict__ z2, const float* __restrict__ dinv,
                            float* __restrict__ acc2, int n) {
    int i = blockIdx.x * blockDim.x + threadIdx.x;   // one float4 per thread
    if (i < n * (GH / 4)) {
        int node = i / (GH / 4);
        float d = dinv[node];
        float4 v = ((const float4*)z2)[i];
        float4 r; r.x = d * v.x; r.y = d * v.y; r.z = d * v.z; r.w = d * v.w;
        ((float4*)acc2)[i] = r;
    }
}

// acc2[dst] += dinv[src]*z2[src]   (32 threads per edge, 4 channels each)
__global__ void k_scatter2(const int* __restrict__ src, const int* __restrict__ dst,
                           const float* __restrict__ dinv, const float* __restrict__ z2,
                           float* __restrict__ acc2, int E) {
    long long t = (long long)blockIdx.x * blockDim.x + threadIdx.x;
    if (t < (long long)E * 32) {
        int e = (int)(t >> 5);
        int c = ((int)t & 31) * 4;
        int sn = src[e], dn = dst[e];
        float d = dinv[sn];
        const float4 v = *(const float4*)(z2 + (size_t)sn * GH + c);
        float* p = acc2 + (size_t)dn * GH + c;
        atomicAdd(p + 0, d * v.x);
        atomicAdd(p + 1, d * v.y);
        atomicAdd(p + 2, d * v.z);
        atomicAdd(p + 3, d * v.w);
    }
}

// h2[n][j] = relu(dinv[n]*acc2[n][j] + b2[j]) -> f16
__global__ void k_h2(const float* __restrict__ acc2, const float* __restrict__ dinv,
                     const float* __restrict__ b2, _Float16* __restrict__ h2h, int n) {
    int i = blockIdx.x * blockDim.x + threadIdx.x;
    if (i < n * GH) {
        int node = i >> 7, j = i & 127;
        float v = dinv[node] * acc2[i] + b2[j];
        h2h[i] = (_Float16)(v > 0.0f ? v : 0.0f);
    }
}

// ---------------------------------------------------------------------------
// WMMA fragment helpers (v_wmma_f32_16x16x32_f16 layouts per CDNA5 ISA 7.12.2)
// ---------------------------------------------------------------------------

// A: row-major M x K f16. lane = 16*h + m_low; halves 0-7 <- K = k0+8h.. ,
// halves 8-15 <- K = k0+16+8h..
__device__ __forceinline__ v16h load_a_frag(const _Float16* A, int lda,
                                            int m0, int k0, int lane) {
    int m = m0 + (lane & 15);
    int h = lane >> 4;
    const _Float16* p = A + (size_t)m * lda + k0 + 8 * h;
    union { v16h v; v8h q[2]; } u;
    u.q[0] = *(const v8h*)(p);
    u.q[1] = *(const v8h*)(p + 16);
    return u.v;
}

// same fragment layout, reading a padded tile held in LDS (emits ds_load_b128)
__device__ __forceinline__ v16h load_a_frag_lds(const _Float16* sA,
                                                int mloc, int k0, int lane) {
    int m = mloc + (lane & 15);
    int h = lane >> 4;
    const _Float16* p = sA + m * LDS_LDA + k0 + 8 * h;
    union { v16h v; v8h q[2]; } u;
    u.q[0] = *(const v8h*)(p);
    u.q[1] = *(const v8h*)(p + 16);
    return u.v;
}

// B loaded from row-major N x K operand (i.e. B^T): lane holds column n = n0+(lane&15),
// halves 0-7 <- K = k0+8h.., halves 8-15 <- K = k0+16+8h.. (h = lane>>4)
__device__ __forceinline__ v16h load_b_frag(const _Float16* Bt, int ldb,
                                            int n0, int k0, int lane) {
    int n = n0 + (lane & 15);
    int h = lane >> 4;
    const _Float16* p = Bt + (size_t)n * ldb + k0 + 8 * h;
    union { v16h v; v8h q[2]; } u;
    u.q[0] = *(const v8h*)(p);
    u.q[1] = *(const v8h*)(p + 16);
    return u.v;
}

#define WMMA_F16(a, b, c) \
    __builtin_amdgcn_wmma_f32_16x16x32_f16(false, (a), false, (b), (short)0, (c), false, false)

// ---------------------------------------------------------------------------
// GEMM 1: z2 (M x 128, f32) = h1h (M x 128 f16) @ W2t^T (128 x 128)
// one 16x16 tile per wave, 8 waves per block
// ---------------------------------------------------------------------------
__global__ void gemm_z2(const _Float16* __restrict__ A, const _Float16* __restrict__ Bt,
                        float* __restrict__ C, int M) {
    int wave = threadIdx.x >> 5;
    int lane = threadIdx.x & 31;
    int tile = blockIdx.x * 8 + wave;
    const int tiles_n = GH / 16;                 // 8
    int tm = tile / tiles_n, tn = tile % tiles_n;
    if (tm * 16 >= M) return;
    v8f acc = {};
#pragma unroll
    for (int k0 = 0; k0 < GH; k0 += 32) {
        v16h a = load_a_frag(A, GH, tm * 16, k0, lane);
        v16h b = load_b_frag(Bt, GH, tn * 16, k0, lane);
        acc = WMMA_F16(a, b, acc);
    }
    int n = tn * 16 + (lane & 15);
    int mr = tm * 16 + 8 * (lane >> 4);
#pragma unroll
    for (int r = 0; r < 8; ++r)
        C[(size_t)(mr + r) * GH + n] = acc[r];
}

// ---------------------------------------------------------------------------
// GEMM 2: h3h (M x 128, f16) = relu(h2h @ Wlt^T + bl)
// ---------------------------------------------------------------------------
__global__ void gemm_h3(const _Float16* __restrict__ A, const _Float16* __restrict__ Bt,
                        const float* __restrict__ bias, _Float16* __restrict__ C, int M) {
    int wave = threadIdx.x >> 5;
    int lane = threadIdx.x & 31;
    int tile = blockIdx.x * 8 + wave;
    const int tiles_n = GH / 16;
    int tm = tile / tiles_n, tn = tile % tiles_n;
    if (tm * 16 >= M) return;
    v8f acc = {};
#pragma unroll
    for (int k0 = 0; k0 < GH; k0 += 32) {
        v16h a = load_a_frag(A, GH, tm * 16, k0, lane);
        v16h b = load_b_frag(Bt, GH, tn * 16, k0, lane);
        acc = WMMA_F16(a, b, acc);
    }
    int n = tn * 16 + (lane & 15);
    int mr = tm * 16 + 8 * (lane >> 4);
    float bv = bias[n];
#pragma unroll
    for (int r = 0; r < 8; ++r) {
        float v = acc[r] + bv;
        C[(size_t)(mr + r) * GH + n] = (_Float16)(v > 0.0f ? v : 0.0f);
    }
}

// ---------------------------------------------------------------------------
// GEMM 3: out (4096 x 16384, f32) = clip(users @ items^T, 1, 5)
// Block = 8 waves sharing one 32-row A tile staged to LDS via
// global_load_async_to_lds_b128 (ASYNCcnt); each wave owns a 32x32 macro-tile
// (2x2 fragment reuse -> 16 v_wmma per wave). A frags come from LDS (ds_load).
// ---------------------------------------------------------------------------
__global__ void gemm_scores(const _Float16* __restrict__ U, const _Float16* __restrict__ I,
                            float* __restrict__ out) {
    __shared__ _Float16 sA[32 * LDS_LDA];        // padded 32 x 128 tile (8.5 KB)
    const int Ntiles = NITEMS / 32;              // 512 (divisible by 8: waves share tm)
    int wave = threadIdx.x >> 5;
    int lane = threadIdx.x & 31;
    int t0 = blockIdx.x * 8;
    int tm = t0 / Ntiles;
    int tn = t0 % Ntiles + wave;
    int m0 = tm * 32, n0 = tn * 32;

    // async-stage A tile: 512 16B chunks, 2 per thread.
    // chunk c -> row = c>>4, 8-half group g = c&15
    {
        int tid = threadIdx.x;
#pragma unroll
        for (int c = tid; c < 512; c += 256) {
            int row = c >> 4, g = c & 15;
            const _Float16* gp = U + (size_t)(m0 + row) * GH + g * 8;
            lds_f16* lp = (lds_f16*)&sA[row * LDS_LDA + g * 8];
            asm volatile("global_load_async_to_lds_b128 %0, %1, off"
                         :: "v"(lp), "v"((unsigned long long)(size_t)gp)
                         : "memory");
        }
        asm volatile("s_wait_asynccnt 0" ::: "memory");
    }
    __syncthreads();

    v8f acc00 = {}, acc01 = {}, acc10 = {}, acc11 = {};
#pragma unroll
    for (int k0 = 0; k0 < GH; k0 += 32) {
        v16h a0 = load_a_frag_lds(sA, 0,  k0, lane);
        v16h a1 = load_a_frag_lds(sA, 16, k0, lane);
        v16h b0 = load_b_frag(I, GH, n0,      k0, lane);
        v16h b1 = load_b_frag(I, GH, n0 + 16, k0, lane);
        acc00 = WMMA_F16(a0, b0, acc00);
        acc01 = WMMA_F16(a0, b1, acc01);
        acc10 = WMMA_F16(a1, b0, acc10);
        acc11 = WMMA_F16(a1, b1, acc11);
    }
    int nc = lane & 15;
    int mr = 8 * (lane >> 4);
#pragma unroll
    for (int r = 0; r < 8; ++r) {
        size_t row0 = (size_t)(m0 + mr + r) * NITEMS;
        size_t row1 = (size_t)(m0 + 16 + mr + r) * NITEMS;
        float v;
        v = acc00[r]; v = fminf(fmaxf(v, 1.0f), 5.0f); out[row0 + n0 + nc]      = v;
        v = acc01[r]; v = fminf(fmaxf(v, 1.0f), 5.0f); out[row0 + n0 + 16 + nc] = v;
        v = acc10[r]; v = fminf(fmaxf(v, 1.0f), 5.0f); out[row1 + n0 + nc]      = v;
        v = acc11[r]; v = fminf(fmaxf(v, 1.0f), 5.0f); out[row1 + n0 + 16 + nc] = v;
    }
}

// ---------------------------------------------------------------------------
// launch
// ---------------------------------------------------------------------------
static inline size_t alignup(size_t x) { return (x + 255) & ~(size_t)255; }

extern "C" void kernel_launch(void* const* d_in, const int* in_sizes, int n_in,
                              void* d_out, int out_size, void* d_ws, size_t ws_size,
                              hipStream_t stream) {
    const int*   x   = (const int*)d_in[0];
    const int*   ei  = (const int*)d_in[1];
    const float* W1  = (const float*)d_in[3];
    const float* b1  = (const float*)d_in[4];
    const float* W2  = (const float*)d_in[5];
    const float* b2  = (const float*)d_in[6];
    const float* Wl  = (const float*)d_in[7];
    const float* bl  = (const float*)d_in[8];
    float*       out = (float*)d_out;

    const int E = in_sizes[1] / 2;
    const int* src = ei;
    const int* dst = ei + E;

    // workspace carve-out
    char* w = (char*)d_ws;
    float* dinv = (float*)w;              w += alignup((size_t)GN * 4);
    float* s    = (float*)w;              w += alignup((size_t)GN * 4);
    _Float16* W2t = (_Float16*)w;         w += alignup((size_t)GH * GH * 2);
    _Float16* Wlt = (_Float16*)w;         w += alignup((size_t)GH * GH * 2);
    _Float16* h1h = (_Float16*)w;         w += alignup((size_t)GN * GH * 2);
    float* z2   = (float*)w;              w += alignup((size_t)GN * GH * 4);
    float* acc2 = (float*)w;              w += alignup((size_t)GN * GH * 4);
    _Float16* h2h = h1h;                  // h1h dead after gemm_z2
    _Float16* h3h = (_Float16*)z2;        // z2 dead after k_h2

    const int T = 256;
    // degree -> dinv
    k_fill1<<<(GN + T - 1) / T, T, 0, stream>>>(dinv, GN);
    k_degcount<<<(E + T - 1) / T, T, 0, stream>>>(dst, dinv, E);
    k_rsqrt_inplace<<<(GN + T - 1) / T, T, 0, stream>>>(dinv, GN);
    // layer 1 (scalar aggregation + rank-1 expansion)
    k_s_init<<<(GN + T - 1) / T, T, 0, stream>>>(x, dinv, s, GN);
    k_s_scatter<<<(E + T - 1) / T, T, 0, stream>>>(src, dst, x, dinv, s, E);
    k_h1<<<((GN * GH) + T - 1) / T, T, 0, stream>>>(s, dinv, W1, b1, h1h, GN);
    // weight transpose/convert
    k_wt<<<((GH * GH) + T - 1) / T, T, 0, stream>>>(W2, Wl, W2t, Wlt);
    // layer 2 linear (WMMA)
    {
        int tiles = (GN / 16) * (GH / 16);
        gemm_z2<<<tiles / 8, 256, 0, stream>>>(h1h, W2t, z2, GN);
    }
    // layer 2 aggregation
    k_acc2_init<<<((GN * GH / 4) + T - 1) / T, T, 0, stream>>>(z2, dinv, acc2, GN);
    {
        long long nthr = (long long)E * 32;
        k_scatter2<<<(unsigned)((nthr + T - 1) / T), T, 0, stream>>>(src, dst, dinv, z2, acc2, E);
    }
    k_h2<<<((GN * GH) + T - 1) / T, T, 0, stream>>>(acc2, dinv, b2, h2h, GN);
    // final linear + relu (WMMA)
    {
        int tiles = (GN / 16) * (GH / 16);
        gemm_h3<<<tiles / 8, 256, 0, stream>>>(h2h, Wlt, bl, h3h, GN);
    }
    // scoring GEMM (WMMA + async-LDS A staging) + clip
    {
        int waves = (NUSERS / 32) * (NITEMS / 32);   // 65536
        gemm_scores<<<waves / 8, 256, 0, stream>>>(h3h, h3h + (size_t)NUSERS * GH, out);
    }
}
